// CrossModalAttention2_57449482551604
// MI455X (gfx1250) — matrix operand order repaired and checked by reading, not compiled, not run on hardware
//
#include <hip/hip_runtime.h>
#include <hip/hip_bf16.h>
#include <math.h>

typedef __attribute__((ext_vector_type(16))) __bf16 v16bf;
typedef __attribute__((ext_vector_type(8)))  float  v8f;

namespace {
constexpr int kB = 4;
constexpr int kC = 128;
constexpr int kN = 4096;   // H*W
constexpr int kM = 2048;   // N_t
constexpr float kScale = 0.015625f; // 1/sqrt(N) = 1/64
}

union BF16x16 { v16bf v; uint4 q[2]; };
union F4x2    { float4 v4[2]; float f[8]; };

__device__ __forceinline__ v8f wmma_bf16(v16bf a, v16bf b, v8f c) {
  // D = A(16x32 bf16) * B(32x16 bf16) + C(16x16 f32)
  return __builtin_amdgcn_wmma_f32_16x16x32_bf16(false, a, false, b, (short)0, c,
                                                 false, false);
}

// ---------------- prep: fp32 -> bf16 (direct) ----------------
__global__ void cvt_bf16_kernel(const float* __restrict__ in, __bf16* __restrict__ out, int n) {
  int i = blockIdx.x * 256 + threadIdx.x;
  if (i < n) out[i] = (__bf16)in[i];
}

// ---------------- prep: fp32 [rows,cols] -> bf16 [cols,rows] (LDS tiled) ----------------
__global__ void transpose_cvt_kernel(const float* __restrict__ in, __bf16* __restrict__ out,
                                     int rows, int cols) {
  __shared__ float tile[32][33];
  const size_t bo = (size_t)blockIdx.z * rows * cols;
  in += bo; out += bo;
  const int x = blockIdx.x * 32 + threadIdx.x;      // input col
  const int y = blockIdx.y * 32 + threadIdx.y;      // input row base
#pragma unroll
  for (int j = 0; j < 4; ++j)
    tile[threadIdx.y + j * 8][threadIdx.x] = in[(size_t)(y + j * 8) * cols + x];
  __syncthreads();
  const int ox = blockIdx.y * 32 + threadIdx.x;     // output col (= input row)
  const int oy = blockIdx.x * 32 + threadIdx.y;     // output row (= input col)
#pragma unroll
  for (int j = 0; j < 4; ++j)
    out[(size_t)(oy + j * 8) * rows + ox] = (__bf16)tile[threadIdx.x][threadIdx.y + j * 8];
}

// ---------------- GEMM1: S[n,m] = scale * sum_c Q[n,c] * T[m,c] ----------------
// grid: (kM/128, kN/128, kB), block 256 (8 waves). Wave tile: 32n x 64m.
__global__ void gemm1_kernel(const __bf16* __restrict__ Q, const __bf16* __restrict__ T,
                             float* __restrict__ S) {
  const int lane  = threadIdx.x & 31;
  const int wave  = threadIdx.x >> 5;
  const int lrow  = lane & 15;
  const int lhalf = lane >> 4;
  const int b     = blockIdx.z;
  const int nOff  = blockIdx.y * 128 + (wave & 3) * 32;
  const int mOff  = blockIdx.x * 128 + (wave >> 2) * 64;
  Q += (size_t)b * kN * kC;
  T += (size_t)b * kM * kC;
  S += (size_t)b * kN * kM;

  v8f acc[2][4];
#pragma unroll
  for (int i = 0; i < 2; ++i)
#pragma unroll
    for (int j = 0; j < 4; ++j) acc[i][j] = (v8f){0.f,0.f,0.f,0.f,0.f,0.f,0.f,0.f};

  for (int c0 = 0; c0 < kC; c0 += 32) {
    BF16x16 a[2];
#pragma unroll
    for (int ns = 0; ns < 2; ++ns) {
      // A frag (16x32): lane L -> row lrow, K = lhalf*8 + {0..7} and +16
      const __bf16* ap = Q + (size_t)(nOff + ns * 16 + lrow) * kC + c0 + lhalf * 8;
      a[ns].q[0] = *(const uint4*)ap;
      a[ns].q[1] = *(const uint4*)(ap + 16);
    }
    BF16x16 bm[4];
#pragma unroll
    for (int ms = 0; ms < 4; ++ms) {
      // B frag (32x16): lane L -> col lrow, K = lhalf*8 + {0..7} and +16; B[k,col]=T[col, c0+k]
      const __bf16* bp = T + (size_t)(mOff + ms * 16 + lrow) * kC + c0 + lhalf * 8;
      bm[ms].q[0] = *(const uint4*)bp;
      bm[ms].q[1] = *(const uint4*)(bp + 16);
    }
#pragma unroll
    for (int ns = 0; ns < 2; ++ns)
#pragma unroll
      for (int ms = 0; ms < 4; ++ms)
        acc[ns][ms] = wmma_bf16(a[ns].v, bm[ms].v, acc[ns][ms]);
  }
#pragma unroll
  for (int ns = 0; ns < 2; ++ns) {
#pragma unroll
    for (int ms = 0; ms < 4; ++ms) {
      const int mcol = mOff + ms * 16 + lrow;
#pragma unroll
      for (int i = 0; i < 8; ++i) {
        const int nrow = nOff + ns * 16 + lhalf * 8 + i;
        S[(size_t)nrow * kM + mcol] = acc[ns][ms][i] * kScale;
      }
    }
  }
}

// ---------------- softmax stats over the n (query) axis, per column m ----------------
__global__ void colmax_part_kernel(const float* __restrict__ S, float* __restrict__ part) {
  const int m = blockIdx.x * 256 + threadIdx.x;
  const int chunk = blockIdx.y;
  const int b = blockIdx.z;
  const float* s = S + (size_t)b * kN * kM + (size_t)chunk * 512 * kM + m;
  float mx = -3.4e38f;
  for (int i = 0; i < 512; ++i) mx = fmaxf(mx, s[(size_t)i * kM]);
  part[((size_t)b * 8 + chunk) * kM + m] = mx;
}

__global__ void colmax_reduce_kernel(const float* __restrict__ part, float* __restrict__ mx) {
  const int m = blockIdx.x * 256 + threadIdx.x;
  const int b = blockIdx.y;
  float v = -3.4e38f;
#pragma unroll
  for (int i = 0; i < 8; ++i) v = fmaxf(v, part[((size_t)b * 8 + i) * kM + m]);
  mx[(size_t)b * kM + m] = v;
}

__global__ void colsum_part_kernel(const float* __restrict__ S, const float* __restrict__ mx,
                                   float* __restrict__ part) {
  const int m = blockIdx.x * 256 + threadIdx.x;
  const int chunk = blockIdx.y;
  const int b = blockIdx.z;
  const float cmax = mx[(size_t)b * kM + m];
  const float* s = S + (size_t)b * kN * kM + (size_t)chunk * 512 * kM + m;
  float acc = 0.f;
  for (int i = 0; i < 512; ++i) acc += __expf(s[(size_t)i * kM] - cmax);
  part[((size_t)b * 8 + chunk) * kM + m] = acc;
}

__global__ void colsum_reduce_kernel(const float* __restrict__ part, float* __restrict__ rcp) {
  const int m = blockIdx.x * 256 + threadIdx.x;
  const int b = blockIdx.y;
  float v = 0.f;
#pragma unroll
  for (int i = 0; i < 8; ++i) v += part[((size_t)b * 8 + i) * kM + m];
  rcp[(size_t)b * kM + m] = 1.0f / v;
}

// ---------------- GEMM2b: out1[b][c][n] = Fs[b][c][n] + sum_m P[n,m]*Ft[m,c] ----------------
// grid: (kN/128, kB), block 256. Wave: 16n x 128c, K loop over m.
__global__ void gemm2b_kernel(const float* __restrict__ S, const float* __restrict__ mx,
                              const float* __restrict__ rcp, const __bf16* __restrict__ Tt,
                              const float* __restrict__ Fs, float* __restrict__ out1) {
  const int lane  = threadIdx.x & 31;
  const int wave  = threadIdx.x >> 5;
  const int lrow  = lane & 15;
  const int lhalf = lane >> 4;
  const int b     = blockIdx.y;
  const int nRowA = blockIdx.x * 128 + wave * 16 + lrow;   // A-matrix row for this lane
  const float* srow = S + ((size_t)b * kN + nRowA) * kM;
  const float* mrow = mx + (size_t)b * kM;
  const float* rrow = rcp + (size_t)b * kM;
  Tt += (size_t)b * kC * kM;
  const int h8 = lhalf * 8;

  v8f acc[8];
#pragma unroll
  for (int i = 0; i < 8; ++i) acc[i] = (v8f){0.f,0.f,0.f,0.f,0.f,0.f,0.f,0.f};

  for (int m0 = 0; m0 < kM; m0 += 32) {
    F4x2 sv0, sv1, xv0, xv1, rv0, rv1;
    sv0.v4[0] = *(const float4*)(srow + m0 + h8);
    sv0.v4[1] = *(const float4*)(srow + m0 + h8 + 4);
    sv1.v4[0] = *(const float4*)(srow + m0 + 16 + h8);
    sv1.v4[1] = *(const float4*)(srow + m0 + 16 + h8 + 4);
    xv0.v4[0] = *(const float4*)(mrow + m0 + h8);
    xv0.v4[1] = *(const float4*)(mrow + m0 + h8 + 4);
    xv1.v4[0] = *(const float4*)(mrow + m0 + 16 + h8);
    xv1.v4[1] = *(const float4*)(mrow + m0 + 16 + h8 + 4);
    rv0.v4[0] = *(const float4*)(rrow + m0 + h8);
    rv0.v4[1] = *(const float4*)(rrow + m0 + h8 + 4);
    rv1.v4[0] = *(const float4*)(rrow + m0 + 16 + h8);
    rv1.v4[1] = *(const float4*)(rrow + m0 + 16 + h8 + 4);

    v16bf a;
#pragma unroll
    for (int e = 0; e < 8; ++e) {
      a[e]     = (__bf16)(__expf(sv0.f[e] - xv0.f[e]) * rv0.f[e]);
      a[e + 8] = (__bf16)(__expf(sv1.f[e] - xv1.f[e]) * rv1.f[e]);
    }
#pragma unroll
    for (int cs = 0; cs < 8; ++cs) {
      BF16x16 bm;
      const __bf16* bp = Tt + (size_t)(cs * 16 + lrow) * kM + m0 + lhalf * 8;
      bm.q[0] = *(const uint4*)bp;
      bm.q[1] = *(const uint4*)(bp + 16);
      acc[cs] = wmma_bf16(a, bm.v, acc[cs]);
    }
  }
  // epilogue: transposed store into [B,C,H,W] with residual add
  const int nBase = blockIdx.x * 128 + wave * 16 + h8;
#pragma unroll
  for (int cs = 0; cs < 8; ++cs) {
    const int c = cs * 16 + lrow;
    const float* fp = Fs + ((size_t)b * kC + c) * kN + nBase;
    float* op = out1 + ((size_t)b * kC + c) * kN + nBase;
    float4 f0 = *(const float4*)fp;
    float4 f1 = *(const float4*)(fp + 4);
    float4 o0, o1;
    o0.x = f0.x + acc[cs][0]; o0.y = f0.y + acc[cs][1];
    o0.z = f0.z + acc[cs][2]; o0.w = f0.w + acc[cs][3];
    o1.x = f1.x + acc[cs][4]; o1.y = f1.y + acc[cs][5];
    o1.z = f1.z + acc[cs][6]; o1.w = f1.w + acc[cs][7];
    *(float4*)op = o0;
    *(float4*)(op + 4) = o1;
  }
}

// ---------------- GEMM2a: out2[b][m][c] = sum_n P[n,m]*Q[n,c] (A = P^T) ----------------
// grid: (kM/64, kB), block 256. Wave: 16m x 64c, K loop over n.
__global__ void gemm2a_kernel(const float* __restrict__ S, const float* __restrict__ mx,
                              const float* __restrict__ rcp, const __bf16* __restrict__ Qt,
                              float* __restrict__ out2) {
  const int lane  = threadIdx.x & 31;
  const int wave  = threadIdx.x >> 5;
  const int lrow  = lane & 15;
  const int lhalf = lane >> 4;
  const int b     = blockIdx.y;
  const int mSub  = blockIdx.x * 64 + (wave & 3) * 16;
  const int cBase = (wave >> 2) * 64;
  const int mcol  = mSub + lrow;                 // A row (token index) for this lane
  const float cmax = mx[(size_t)b * kM + mcol];
  const float crcp = rcp[(size_t)b * kM + mcol];
  const float* sb = S + (size_t)b * kN * kM + mcol;
  Qt += (size_t)b * kC * kN;
  const int h8 = lhalf * 8;

  v8f acc[4];
#pragma unroll
  for (int i = 0; i < 4; ++i) acc[i] = (v8f){0.f,0.f,0.f,0.f,0.f,0.f,0.f,0.f};

  for (int n0 = 0; n0 < kN; n0 += 32) {
    // A = P^T frag: lane row = m, K(n) = n0 + h8 + {0..7} and n0 + 16 + h8 + {0..7}
    const float* sp0 = sb + (size_t)(n0 + h8) * kM;
    const float* sp1 = sb + (size_t)(n0 + 16 + h8) * kM;
    v16bf a;
#pragma unroll
    for (int e = 0; e < 8; ++e) {
      a[e]     = (__bf16)(__expf(sp0[(size_t)e * kM] - cmax) * crcp);
      a[e + 8] = (__bf16)(__expf(sp1[(size_t)e * kM] - cmax) * crcp);
    }
#pragma unroll
    for (int cs = 0; cs < 4; ++cs) {
      BF16x16 bm;  // B[k=n, col=c] = Qt[c][n0+k]  (contiguous in n)
      const __bf16* bp = Qt + (size_t)(cBase + cs * 16 + lrow) * kN + n0 + lhalf * 8;
      bm.q[0] = *(const uint4*)bp;
      bm.q[1] = *(const uint4*)(bp + 16);
      acc[cs] = wmma_bf16(a, bm.v, acc[cs]);
    }
  }
#pragma unroll
  for (int cs = 0; cs < 4; ++cs) {
    const int c = cBase + cs * 16 + lrow;
#pragma unroll
    for (int i = 0; i < 8; ++i) {
      const int mrow = mSub + h8 + i;
      out2[((size_t)b * kM + mrow) * kC + c] = acc[cs][i];
    }
  }
}

extern "C" void kernel_launch(void* const* d_in, const int* in_sizes, int n_in,
                              void* d_out, int out_size, void* d_ws, size_t ws_size,
                              hipStream_t stream) {
  (void)in_sizes; (void)n_in; (void)out_size; (void)ws_size;
  const float* Fs = (const float*)d_in[0];   // [B, C, H, W] = [B, C, N]
  const float* Ft = (const float*)d_in[1];   // [B, M, C]
  float* out = (float*)d_out;                // F_s_updated (B*C*N) then F_t_updated (B*M*C)

  char* ws = (char*)d_ws;
  size_t off = 0;
  __bf16* Q   = (__bf16*)(ws + off); off += (size_t)kB * kN * kC * 2;  // [B,N,C]
  __bf16* Qt  = (__bf16*)(ws + off); off += (size_t)kB * kN * kC * 2;  // [B,C,N]
  __bf16* T   = (__bf16*)(ws + off); off += (size_t)kB * kM * kC * 2;  // [B,M,C]
  __bf16* Tt  = (__bf16*)(ws + off); off += (size_t)kB * kM * kC * 2;  // [B,C,M]
  float*  S   = (float*) (ws + off); off += (size_t)kB * kN * kM * 4;  // scores (scaled)
  float*  maxP= (float*) (ws + off); off += (size_t)kB * 8 * kM * 4;
  float*  sumP= (float*) (ws + off); off += (size_t)kB * 8 * kM * 4;
  float*  cmax= (float*) (ws + off); off += (size_t)kB * kM * 4;
  float*  crcp= (float*) (ws + off); off += (size_t)kB * kM * 4;

  // prep
  cvt_bf16_kernel<<<(kB * kC * kN + 255) / 256, 256, 0, stream>>>(Fs, Qt, kB * kC * kN);
  cvt_bf16_kernel<<<(kB * kM * kC + 255) / 256, 256, 0, stream>>>(Ft, T, kB * kM * kC);
  transpose_cvt_kernel<<<dim3(kN / 32, kC / 32, kB), dim3(32, 8), 0, stream>>>(Fs, Q, kC, kN);
  transpose_cvt_kernel<<<dim3(kC / 32, kM / 32, kB), dim3(32, 8), 0, stream>>>(Ft, Tt, kM, kC);
  // scores
  gemm1_kernel<<<dim3(kM / 128, kN / 128, kB), 256, 0, stream>>>(Q, T, S);
  // softmax stats (over n, per column m)
  colmax_part_kernel<<<dim3(kM / 256, 8, kB), 256, 0, stream>>>(S, maxP);
  colmax_reduce_kernel<<<dim3(kM / 256, kB), 256, 0, stream>>>(maxP, cmax);
  colsum_part_kernel<<<dim3(kM / 256, 8, kB), 256, 0, stream>>>(S, cmax, sumP);
  colsum_reduce_kernel<<<dim3(kM / 256, kB), 256, 0, stream>>>(sumP, crcp);
  // consumers
  gemm2b_kernel<<<dim3(kN / 128, kB), 256, 0, stream>>>(S, cmax, crcp, Tt, Fs, out);
  gemm2a_kernel<<<dim3(kM / 64, kB), 256, 0, stream>>>(S, cmax, crcp, Qt,
                                                       out + (size_t)kB * kC * kN);
}
